// MambaSparseMoeBlock_33354716021469
// MI455X (gfx1250) — matrix operand order repaired and checked by reading, not compiled, not run on hardware
//
#include <hip/hip_runtime.h>
#include <hip/hip_bf16.h>
#include <math.h>

// ---------------- problem constants ----------------
#define NUM_EXPERTS 8
#define TOP_K 2
#define D_MODEL 768
#define D_FF 2048
#define T_TOKENS 4096   // B*S = 2*2048

typedef __attribute__((ext_vector_type(16))) __bf16 v16bf;
typedef __attribute__((ext_vector_type(8)))  __bf16 v8bf;
typedef __attribute__((ext_vector_type(8)))  float  v8f;

// ---------------------------------------------------------------------------
// fp32 -> bf16 vectorized cast: 32B loads, 16B stores (HBM-bound phase)
// ---------------------------------------------------------------------------
__global__ void cast_f32_bf16(const float* __restrict__ src,
                              __bf16* __restrict__ dst, size_t n8) {
  size_t i = (size_t)blockIdx.x * blockDim.x + threadIdx.x;
  size_t stride = (size_t)gridDim.x * blockDim.x;
  const v8f* s = (const v8f*)src;
  v8bf* d = (v8bf*)dst;
  for (; i < n8; i += stride) {
    v8f x = s[i];
    v8bf y;
#pragma unroll
    for (int j = 0; j < 8; ++j) y[j] = (__bf16)x[j];
    d[i] = y;
  }
}

// ---------------------------------------------------------------------------
// Router: one wave32 per token. Lane-split dot products over D_MODEL,
// xor-shuffle reduce, lane 0 does softmax + top-2 + renormalize.
// ---------------------------------------------------------------------------
__global__ void router_kernel(const float* __restrict__ X,
                              const float* __restrict__ GW,
                              float* __restrict__ logits_out,
                              float* __restrict__ comb) {
  const int lane = threadIdx.x & 31;
  const int wavesPerBlock = blockDim.x >> 5;
  const int t = blockIdx.x * wavesPerBlock + (threadIdx.x >> 5);
  if (t >= T_TOKENS) return;

  float p[NUM_EXPERTS];
#pragma unroll
  for (int e = 0; e < NUM_EXPERTS; ++e) p[e] = 0.0f;

  const float* xrow = X + (size_t)t * D_MODEL;
  for (int k = lane; k < D_MODEL; k += 32) {
    float xv = xrow[k];
#pragma unroll
    for (int e = 0; e < NUM_EXPERTS; ++e)
      p[e] += xv * GW[(size_t)e * D_MODEL + k];
  }
#pragma unroll
  for (int e = 0; e < NUM_EXPERTS; ++e) {
#pragma unroll
    for (int off = 16; off > 0; off >>= 1)
      p[e] += __shfl_xor(p[e], off, 32);
  }

  if (lane == 0) {
#pragma unroll
    for (int e = 0; e < NUM_EXPERTS; ++e)
      logits_out[(size_t)t * NUM_EXPERTS + e] = p[e];

    float m = p[0];
#pragma unroll
    for (int e = 1; e < NUM_EXPERTS; ++e) m = fmaxf(m, p[e]);
    float prob[NUM_EXPERTS];
    float se = 0.0f;
#pragma unroll
    for (int e = 0; e < NUM_EXPERTS; ++e) { prob[e] = __expf(p[e] - m); se += prob[e]; }
    float inv = 1.0f / se;
#pragma unroll
    for (int e = 0; e < NUM_EXPERTS; ++e) prob[e] *= inv;

    int i0 = 0;
#pragma unroll
    for (int e = 1; e < NUM_EXPERTS; ++e) if (prob[e] > prob[i0]) i0 = e;
    int i1 = (i0 == 0) ? 1 : 0;
#pragma unroll
    for (int e = 0; e < NUM_EXPERTS; ++e)
      if (e != i0 && prob[e] > prob[i1]) i1 = e;

    float w0 = prob[i0], w1 = prob[i1];
    float wsn = 1.0f / (w0 + w1);
    w0 *= wsn; w1 *= wsn;
#pragma unroll
    for (int e = 0; e < NUM_EXPERTS; ++e)
      comb[(size_t)t * NUM_EXPERTS + e] = 0.0f;
    comb[(size_t)t * NUM_EXPERTS + i0] = w0;
    comb[(size_t)t * NUM_EXPERTS + i1] = w1;
  }
}

// ---------------------------------------------------------------------------
// Fragment loaders: contiguous k order, same in A and B (not the ISA table order).
//   A 16x32 bf16: lanes 0-15: M=lane, K {k0..+15};
//                 lanes 16-31: M=lane-16, K {k0+16..+31}.
//   B 32x16 bf16: lanes 0-15: N=lane, K {k0..+15};
//                 lanes 16-31: N=lane-16, K {k0+16..+31}.
// ---------------------------------------------------------------------------
__device__ __forceinline__ v16bf frag_a(const __bf16* p) {
  v8bf lo = *(const v8bf*)(p);
  v8bf hi = *(const v8bf*)(p + 8);
  return __builtin_shufflevector(lo, hi, 0,1,2,3,4,5,6,7,8,9,10,11,12,13,14,15);
}
__device__ __forceinline__ v16bf frag_b(const __bf16* p) {
  v8bf lo = *(const v8bf*)(p);
  v8bf hi = *(const v8bf*)(p + 8);
  return __builtin_shufflevector(lo, hi, 0,1,2,3,4,5,6,7,8,9,10,11,12,13,14,15);
}

#define WMMA_BF16(ACC, A, B) \
  __builtin_amdgcn_wmma_f32_16x16x32_bf16(false, (A), false, (B), (short)0, (ACC), false, false)

// ---------------------------------------------------------------------------
// GEMM1 (per expert): h = comb[:,e] * silu(x@w1^T) * (x@w3^T), bf16 out.
// Wave tile: 64 rows x 32 cols, dual weights -> M4 x N2 x {w1,w3} = 16 WMMAs
// per K-step of 32 against 8 fragment loads (8 KB) -> 32 FLOP/B from L2.
// __launch_bounds__(128, 1) lifts the VGPR budget to 256 so the 16
// accumulators (128 VGPRs) + 8 fragments (64 VGPRs) live in registers
// (round-3 build spilled under the default 128-VGPR occupancy cap).
// ---------------------------------------------------------------------------
__global__ void __launch_bounds__(128, 1)
gemm1_kernel(const __bf16* __restrict__ X,
             const __bf16* __restrict__ W1,
             const __bf16* __restrict__ W3,
             const float*  __restrict__ comb,
             __bf16* __restrict__ H, int expert) {
  const int lane = threadIdx.x & 31;
  const int wave = threadIdx.x >> 5;
  const int l15  = lane & 15;
  const int hsel = (lane >> 4) & 1;
  const int rowBase = blockIdx.y * 128 + (wave >> 1) * 64;
  const int colBase = blockIdx.x * 64  + (wave & 1) * 32;

  const __bf16* w1e = W1 + (size_t)expert * D_FF * D_MODEL;
  const __bf16* w3e = W3 + (size_t)expert * D_FF * D_MODEL;

  const __bf16* aB  = X   + (size_t)(rowBase + l15) * D_MODEL + hsel * 16;
  const __bf16* b1B = w1e + (size_t)(colBase + l15) * D_MODEL + hsel * 16;
  const __bf16* b3B = w3e + (size_t)(colBase + l15) * D_MODEL + hsel * 16;

  v8f acc1[4][2];
  v8f acc3[4][2];
#pragma unroll
  for (int m = 0; m < 4; ++m)
#pragma unroll
    for (int n = 0; n < 2; ++n) { acc1[m][n] = v8f{}; acc3[m][n] = v8f{}; }

#pragma unroll
  for (int k0 = 0; k0 < D_MODEL; k0 += 32) {
    v16bf a[4], b1[2], b3[2];
#pragma unroll
    for (int m = 0; m < 4; ++m)
      a[m] = frag_a(aB + m * 16 * D_MODEL + k0);
#pragma unroll
    for (int n = 0; n < 2; ++n) {
      b1[n] = frag_b(b1B + n * 16 * D_MODEL + k0);
      b3[n] = frag_b(b3B + n * 16 * D_MODEL + k0);
    }
#pragma unroll
    for (int m = 0; m < 4; ++m)
#pragma unroll
      for (int n = 0; n < 2; ++n) {
        acc1[m][n] = WMMA_BF16(acc1[m][n], a[m], b1[n]);
        acc3[m][n] = WMMA_BF16(acc3[m][n], a[m], b3[n]);
      }
  }

  // C/D layout: lanes 0-15 VGPR v = row +v, lanes 16-31 = row +8+v; col = l15.
#pragma unroll
  for (int m = 0; m < 4; ++m) {
    const int roff = rowBase + m * 16 + hsel * 8;
    float cw[8];
#pragma unroll
    for (int v = 0; v < 8; ++v)
      cw[v] = comb[(size_t)(roff + v) * NUM_EXPERTS + expert];
#pragma unroll
    for (int n = 0; n < 2; ++n) {
      const int col = colBase + n * 16 + l15;
#pragma unroll
      for (int v = 0; v < 8; ++v) {
        float g  = acc1[m][n][v];
        float sg = g / (1.0f + __expf(-g));       // silu
        float hv = sg * acc3[m][n][v] * cw[v];
        H[(size_t)(roff + v) * D_FF + col] = (__bf16)hv;
      }
    }
  }
}

// ---------------------------------------------------------------------------
// GEMM2 (per expert): out (+)= h @ w2^T, fp32 out. Wave tile 32x64 ->
// M2 x N4 = 8 WMMAs per K-step. With the 256-VGPR budget the scheduler can
// keep 2 K-iterations of fragments in flight (64 acc + 2x48 = 160 VGPRs).
// Sequential per-expert launches keep fp accumulation order deterministic.
// ---------------------------------------------------------------------------
__global__ void __launch_bounds__(128, 1)
gemm2_kernel(const __bf16* __restrict__ H,
             const __bf16* __restrict__ W2,
             float* __restrict__ out,
             int expert, int accumulate) {
  const int lane = threadIdx.x & 31;
  const int wave = threadIdx.x >> 5;
  const int l15  = lane & 15;
  const int hsel = (lane >> 4) & 1;
  const int rowBase = blockIdx.y * 64  + (wave >> 1) * 32;
  const int colBase = blockIdx.x * 128 + (wave & 1) * 64;

  const __bf16* w2e = W2 + (size_t)expert * D_MODEL * D_FF;

  const __bf16* aB = H   + (size_t)(rowBase + l15) * D_FF + hsel * 16;
  const __bf16* bB = w2e + (size_t)(colBase + l15) * D_FF + hsel * 16;

  v8f acc[2][4] = {{v8f{}, v8f{}, v8f{}, v8f{}}, {v8f{}, v8f{}, v8f{}, v8f{}}};

#pragma unroll
  for (int k0 = 0; k0 < D_FF; k0 += 32) {
    v16bf a[2], b[4];
#pragma unroll
    for (int m = 0; m < 2; ++m)
      a[m] = frag_a(aB + m * 16 * D_FF + k0);
#pragma unroll
    for (int n = 0; n < 4; ++n)
      b[n] = frag_b(bB + n * 16 * D_FF + k0);
#pragma unroll
    for (int m = 0; m < 2; ++m)
#pragma unroll
      for (int n = 0; n < 4; ++n)
        acc[m][n] = WMMA_BF16(acc[m][n], a[m], b[n]);
  }

#pragma unroll
  for (int m = 0; m < 2; ++m) {
    const int roff = rowBase + m * 16 + hsel * 8;
#pragma unroll
    for (int n = 0; n < 4; ++n) {
      const int col = colBase + n * 16 + l15;
#pragma unroll
      for (int v = 0; v < 8; ++v) {
        size_t idx = (size_t)(roff + v) * D_MODEL + col;
        float r = acc[m][n][v];
        out[idx] = accumulate ? (out[idx] + r) : r;
      }
    }
  }
}

// ---------------------------------------------------------------------------
// Host-side launch
// ---------------------------------------------------------------------------
extern "C" void kernel_launch(void* const* d_in, const int* in_sizes, int n_in,
                              void* d_out, int out_size, void* d_ws, size_t ws_size,
                              hipStream_t stream) {
  (void)in_sizes; (void)n_in; (void)out_size; (void)ws_size;

  // setup_inputs() order: hidden_states, gate_w, w1, w2, w3  (all fp32)
  const float* x_f  = (const float*)d_in[0];
  const float* gw_f = (const float*)d_in[1];
  const float* w1_f = (const float*)d_in[2];
  const float* w2_f = (const float*)d_in[3];
  const float* w3_f = (const float*)d_in[4];

  float* out    = (float*)d_out;                       // [T, D_MODEL]
  float* logits = out + (size_t)T_TOKENS * D_MODEL;    // [T, NUM_EXPERTS]

  // --- workspace carve (256B-aligned chunks) ---
  char* ws = (char*)d_ws;
  auto carve = [&](size_t bytes) -> void* {
    void* p = (void*)ws;
    ws += (bytes + 255) & ~(size_t)255;
    return p;
  };
  const size_t nX  = (size_t)T_TOKENS * D_MODEL;
  const size_t nW1 = (size_t)NUM_EXPERTS * D_FF * D_MODEL;
  const size_t nW2 = (size_t)NUM_EXPERTS * D_MODEL * D_FF;
  const size_t nW3 = nW1;
  const size_t nH  = (size_t)T_TOKENS * D_FF;

  __bf16* xb  = (__bf16*)carve(nX * 2);
  __bf16* w1b = (__bf16*)carve(nW1 * 2);
  __bf16* w2b = (__bf16*)carve(nW2 * 2);
  __bf16* w3b = (__bf16*)carve(nW3 * 2);
  __bf16* hb  = (__bf16*)carve(nH * 2);
  float*  comb = (float*)carve((size_t)T_TOKENS * NUM_EXPERTS * 4);

  // --- 1) casts to bf16 (vectorized x8) ---
  cast_f32_bf16<<<1024, 256, 0, stream>>>(x_f,  xb,  nX / 8);
  cast_f32_bf16<<<2048, 256, 0, stream>>>(w1_f, w1b, nW1 / 8);
  cast_f32_bf16<<<2048, 256, 0, stream>>>(w2_f, w2b, nW2 / 8);
  cast_f32_bf16<<<2048, 256, 0, stream>>>(w3_f, w3b, nW3 / 8);

  // --- 2) router: 8 waves/block -> 8 tokens/block ---
  router_kernel<<<T_TOKENS / 8, 256, 0, stream>>>(x_f, gw_f, logits, comb);

  // --- 3) per-expert SwiGLU + down-projection ---
  dim3 g1(D_FF / 64, T_TOKENS / 128);      // (32, 32), 128 thr
  dim3 g2(D_MODEL / 128, T_TOKENS / 64);   // (6, 64),  128 thr
  for (int e = 0; e < NUM_EXPERTS; ++e) {
    gemm1_kernel<<<g1, 128, 0, stream>>>(xb, w1b, w3b, comb, hb, e);
    gemm2_kernel<<<g2, 128, 0, stream>>>(hb, w2b, out, e, (e > 0) ? 1 : 0);
  }
}
